// ClassifierAttn_20040317403771
// MI455X (gfx1250) — matrix operand order repaired and checked by reading, not compiled, not run on hardware
//
#include <hip/hip_runtime.h>
#include <hip/hip_bf16.h>

// ---- problem dims ----
#define NPAIR   80      // B*QPV
#define SEQ_V_  2048
#define SEQ_Q_  32
#define VID_DIM 768
#define HID     512
#define HID2    1024
#define OUT_    5
// ---- tiling ----
#define NSPLIT  8
#define SROWS   256     // SEQ_V_/NSPLIT
#define MT      64      // rows per M tile
#define KC      32      // K per WMMA chunk
#define NKC     24      // VID_DIM/KC
#define VSTR    520     // padded Vs stride (halves)
#define PSTR    520     // partial record stride (floats): [0]=max [1]=l [2..513]=vacc

typedef __attribute__((ext_vector_type(16))) _Float16 v16h;
typedef __attribute__((ext_vector_type(8)))  _Float16 v8h;
typedef __attribute__((ext_vector_type(4)))  _Float16 v4h;
typedef __attribute__((ext_vector_type(8)))  float    v8f;
typedef __attribute__((ext_vector_type(4)))  float    v4f;

// --- gfx1250 async memory->LDS copy (ASYNCcnt-tracked, bypasses VGPRs) ---
__device__ __forceinline__ void async_b128(const _Float16* g, _Float16* l) {
  // generic shared pointer low 32 bits == LDS byte offset (AS3 offset)
  asm volatile("global_load_async_to_lds_b128 %0, %1, off"
               :: "v"((unsigned)(size_t)l), "v"(g) : "memory");
}
__device__ __forceinline__ void wait_async0() {
#if __has_builtin(__builtin_amdgcn_s_wait_asynccnt)
  __builtin_amdgcn_s_wait_asynccnt(0);
#else
  asm volatile("s_wait_asynccnt 0x0" ::: "memory");
#endif
}

// ---------------------------------------------------------------------------
// Kernel A: per-pair q projection, qq = Wq q, folded ck = (Wk^T qq)/sqrt(2H)
// ---------------------------------------------------------------------------
__global__ void ca_setup(const float* __restrict__ ques, const float* __restrict__ Wqp,
                         const float* __restrict__ bqp,  const float* __restrict__ Wq,
                         const float* __restrict__ Wk,
                         float* __restrict__ ws_q, float* __restrict__ ws_ck) {
  __shared__ float quesS[VID_DIM];
  __shared__ float qS[HID];
  __shared__ float qqS[HID2];
  const int t = threadIdx.x, p = blockIdx.x;
  const float* qe = ques + (size_t)p * SEQ_Q_ * VID_DIM;   // [b,qv,0,:]
  for (int i = t; i < VID_DIM; i += 256) quesS[i] = qe[i];
  __syncthreads();
  for (int h = t; h < HID; h += 256) {
    float a = bqp[h];
    const float* w = Wqp + (size_t)h * VID_DIM;
    for (int d = 0; d < VID_DIM; ++d) a += w[d] * quesS[d];
    qS[h] = a;
    ws_q[p * HID + h] = a;
  }
  __syncthreads();
  for (int i = 0; i < 4; ++i) {
    int k = t + i * 256;
    float a = 0.f;
    const float* w = Wq + (size_t)k * HID;
    for (int h = 0; h < HID; ++h) a += w[h] * qS[h];
    qqS[k] = a;
  }
  __syncthreads();
  {
    float a0 = 0.f, a1 = 0.f;
    for (int k = 0; k < HID2; ++k) {
      const float qk = qqS[k];
      const float* row = Wk + (size_t)k * HID;
      a0 += row[t] * qk;
      a1 += row[t + 256] * qk;
    }
    ws_ck[p * HID + t]       = a0 * 0.03125f;   // 1/sqrt(1024)
    ws_ck[p * HID + t + 256] = a1 * 0.03125f;
  }
}

// ---------------------------------------------------------------------------
// Convert Wvp to f16 once (L2-resident, reused by all GEMM blocks)
// ---------------------------------------------------------------------------
__global__ void ca_cvt(const float* __restrict__ Wvp, _Float16* __restrict__ w16) {
  const int i = blockIdx.x * 256 + threadIdx.x;
  if (i < HID * VID_DIM) w16[i] = (_Float16)Wvp[i];
}

// ---------------------------------------------------------------------------
// Kernel B: fused v-projection GEMM (WMMA f16) + scores + online softmax pool
// grid = NPAIR*NSPLIT blocks, 256 threads = 8 wave32
// wave w: M-rows [(w&3)*16, +16), N-cols [(w>>2)*256, +256) of the 64x512 tile
// Double-buffered K pipeline:
//   - async-to-LDS copy of Wvp chunk k+1 issued before the WMMA block
//   - video chunk k+1 global loads issued before WMMAs, cvt+ds_store after
//   - B fragments software-pipelined one tile ahead of each WMMA
// ---------------------------------------------------------------------------
__global__ __launch_bounds__(256, 1)
void ca_attn(const float* __restrict__ video, const _Float16* __restrict__ w16,
             const float* __restrict__ bvp, const float* __restrict__ ws_ck,
             float* __restrict__ ws_part) {
  __shared__ _Float16 As[2][MT * KC];    //  8 KB
  __shared__ _Float16 Bs[2][HID * KC];   // 64 KB
  __shared__ _Float16 Vs[MT * VSTR];     // 65 KB  (v tile, padded)
  __shared__ float ckS[HID];
  __shared__ float vaccS[HID];
  __shared__ float wS[MT];
  __shared__ float redMax[2], redSum[2];
  __shared__ float runMaxS, runLS;

  const int t = threadIdx.x;
  const int lane = t & 31, wv = t >> 5;
  const int pair = blockIdx.x >> 3, split = blockIdx.x & 7;
  const float* vp = video + (size_t)pair * SEQ_V_ * VID_DIM;

  ckS[t]         = ws_ck[pair * HID + t];
  ckS[t + 256]   = ws_ck[pair * HID + t + 256];
  vaccS[t]       = 0.f;
  vaccS[t + 256] = 0.f;
  if (t == 0) { runMaxS = -3.0e38f; runLS = 0.f; }
  __syncthreads();

  const int mbase = (wv & 3) * 16;
  const int nbase = (wv >> 2) * 256;
  const int l15   = lane & 15;
  const int kb8   = (lane >> 4) * 8;    // unused; A frag also uses kb16 (contiguous K)
  const int kb16  = (lane >> 4) * 16;   // A/B-frag K base (contiguous k = 16h + s, not ISA table order)

  // per-thread A-staging coordinates (two 4-float packets per chunk)
  const int ar0 = t >> 3,          ac0 = (t & 7) * 4;
  const int ar1 = (t + 256) >> 3,  ac1 = ((t + 256) & 7) * 4;

  for (int mt = 0; mt < SROWS / MT; ++mt) {
    const int row0 = split * SROWS + mt * MT;
    v8f acc[16];
#pragma unroll
    for (int j = 0; j < 16; ++j)
#pragma unroll
      for (int r = 0; r < 8; ++r) acc[j][r] = 0.f;

    // ---- prologue: stage chunk 0 into buffer 0 (synchronous) ----
    {
#pragma unroll
      for (int i = 0; i < 8; ++i) {
        const int fi = t + i * 256;
        const int h = fi >> 2, c8 = (fi & 3) * 8;
        async_b128(w16 + (size_t)h * VID_DIM + c8, &Bs[0][h * KC + c8]);
      }
      v4f a0 = *(const v4f*)(vp + (size_t)(row0 + ar0) * VID_DIM + ac0);
      v4f a1 = *(const v4f*)(vp + (size_t)(row0 + ar1) * VID_DIM + ac1);
      v4h h0, h1;
      h0[0] = (_Float16)a0[0]; h0[1] = (_Float16)a0[1];
      h0[2] = (_Float16)a0[2]; h0[3] = (_Float16)a0[3];
      h1[0] = (_Float16)a1[0]; h1[1] = (_Float16)a1[1];
      h1[2] = (_Float16)a1[2]; h1[3] = (_Float16)a1[3];
      *(v4h*)&As[0][ar0 * KC + ac0] = h0;
      *(v4h*)&As[0][ar1 * KC + ac1] = h1;
    }
    wait_async0();
    __syncthreads();

    for (int kc = 0; kc < NKC; ++kc) {
      const int cur = kc & 1;
      const bool hn = (kc + 1 < NKC);
      // ---- issue next-chunk staging BEFORE the WMMA block ----
      v4f aR0 = {}, aR1 = {};
      if (hn) {
        const int k0n = (kc + 1) * KC;
#pragma unroll
        for (int i = 0; i < 8; ++i) {
          const int fi = t + i * 256;
          const int h = fi >> 2, c8 = (fi & 3) * 8;
          async_b128(w16 + (size_t)h * VID_DIM + k0n + c8,
                     &Bs[cur ^ 1][h * KC + c8]);
        }
        aR0 = *(const v4f*)(vp + (size_t)(row0 + ar0) * VID_DIM + k0n + ac0);
        aR1 = *(const v4f*)(vp + (size_t)(row0 + ar1) * VID_DIM + k0n + ac1);
        if (kc + 2 < NKC)   // gfx1250 global_prefetch_b8, chunk after next
          __builtin_prefetch(vp + (size_t)(row0 + ar0) * VID_DIM + k0n + KC, 0, 1);
      }

      // ---- A fragment: lane<16 K{0..15}; lane>=16 K{16..31} ----
      v8h alo = *(const v8h*)&As[cur][(mbase + l15) * KC + kb16];
      v8h ahi = *(const v8h*)&As[cur][(mbase + l15) * KC + kb16 + 8];
      v16h afr = __builtin_shufflevector(alo, ahi,
                   0,1,2,3,4,5,6,7,8,9,10,11,12,13,14,15);
      // ---- 16 WMMAs, B fragments software-pipelined one tile ahead ----
      v16h bfr = *(const v16h*)&Bs[cur][(nbase + l15) * KC + kb16];
#pragma unroll
      for (int j = 0; j < 16; ++j) {
        v16h bnx = bfr;
        if (j < 15)
          bnx = *(const v16h*)&Bs[cur][(nbase + (j + 1) * 16 + l15) * KC + kb16];
        acc[j] = __builtin_amdgcn_wmma_f32_16x16x32_f16(
            false, afr, false, bfr, (short)0, acc[j], false, false);
        bfr = bnx;
      }

      // ---- consume A-stage loads AFTER the WMMA block (latency hidden) ----
      if (hn) {
        const int nxt = cur ^ 1;
        v4h h0, h1;
        h0[0] = (_Float16)aR0[0]; h0[1] = (_Float16)aR0[1];
        h0[2] = (_Float16)aR0[2]; h0[3] = (_Float16)aR0[3];
        h1[0] = (_Float16)aR1[0]; h1[1] = (_Float16)aR1[1];
        h1[2] = (_Float16)aR1[2]; h1[3] = (_Float16)aR1[3];
        *(v4h*)&As[nxt][ar0 * KC + ac0] = h0;
        *(v4h*)&As[nxt][ar1 * KC + ac1] = h1;
      }
      wait_async0();        // next B buffer fully landed in LDS
      __syncthreads();      // publish to all waves (also fences ds stores)
    }

    // ---- bias + spill v tile to LDS (C layout: M = r + 8*(lane>=16)) ----
#pragma unroll
    for (int j = 0; j < 16; ++j) {
      const int n = nbase + j * 16 + l15;
      const float bb = bvp[n];
      const int rr = mbase + 8 * (lane >> 4);
#pragma unroll
      for (int r = 0; r < 8; ++r)
        Vs[(rr + r) * VSTR + n] = (_Float16)(acc[j][r] + bb);
    }
    __syncthreads();

    // ---- scores: row dot with ck (waves 0,1 handle 64 rows) ----
    float sc = 0.f;
    if (t < MT) {
      const _Float16* vr = &Vs[t * VSTR];
      for (int n = 0; n < HID; ++n) sc += (float)vr[n] * ckS[n];
      float m = sc;
      for (int off = 16; off; off >>= 1) m = fmaxf(m, __shfl_xor(m, off, 32));
      if (lane == 0) redMax[wv] = m;
    }
    __syncthreads();
    const float tileMax = fmaxf(redMax[0], redMax[1]);
    const float newMax  = fmaxf(runMaxS, tileMax);
    const float factor  = __expf(runMaxS - newMax);
    if (t < MT) {
      const float w = __expf(sc - newMax);
      wS[t] = w;
      float su = w;
      for (int off = 16; off; off >>= 1) su += __shfl_xor(su, off, 32);
      if (lane == 0) redSum[wv] = su;
    }
    __syncthreads();
    const float tileL = redSum[0] + redSum[1];
    // ---- rescale + accumulate weighted rows into vacc (each thr: 2 cols) ----
#pragma unroll
    for (int cc = 0; cc < 2; ++cc) {
      const int c = t + cc * 256;
      float a = vaccS[c] * factor;
      for (int i = 0; i < MT; ++i) a += wS[i] * (float)Vs[i * VSTR + c];
      vaccS[c] = a;
    }
    if (t == 0) { runLS = runLS * factor + tileL; runMaxS = newMax; }
    __syncthreads();
  }

  float* pp = ws_part + (size_t)(pair * NSPLIT + split) * PSTR;
  if (t == 0) { pp[0] = runMaxS; pp[1] = runLS; }
  pp[2 + t]       = vaccS[t];
  pp[2 + t + 256] = vaccS[t + 256];
}

// ---------------------------------------------------------------------------
// Kernel C: combine splits, pooled = vbar*Wv^T, concat q, MLP, masked softmax
// ---------------------------------------------------------------------------
__global__ void ca_final(const float* __restrict__ ws_part, const float* __restrict__ ws_q,
                         const float* __restrict__ Wv, const float* __restrict__ W1,
                         const float* __restrict__ b1, const float* __restrict__ W2,
                         const float* __restrict__ b2, const float* __restrict__ omask,
                         float* __restrict__ out) {
  __shared__ float vbarS[HID];
  __shared__ float xS[HID2];
  __shared__ float h1S[HID];
  __shared__ float logS[OUT_];
  const int t = threadIdx.x, p = blockIdx.x;
  const float* pb = ws_part + (size_t)p * NSPLIT * PSTR;
  float M = -3.0e38f;
  for (int s = 0; s < NSPLIT; ++s) M = fmaxf(M, pb[s * PSTR]);
  float e[NSPLIT];
  float L = 0.f;
  for (int s = 0; s < NSPLIT; ++s) {
    e[s] = __expf(pb[s * PSTR] - M);
    L += pb[s * PSTR + 1] * e[s];
  }
  const float invL = 1.f / L;
#pragma unroll
  for (int cc = 0; cc < 2; ++cc) {
    const int n = t + cc * 256;
    float a = 0.f;
    for (int s = 0; s < NSPLIT; ++s) a += pb[s * PSTR + 2 + n] * e[s];
    vbarS[n] = a * invL;
  }
  __syncthreads();
#pragma unroll
  for (int cc = 0; cc < 2; ++cc) {
    const int h = t + cc * 256;
    const float* w = Wv + (size_t)h * HID;
    float a = 0.f;
    for (int n = 0; n < HID; ++n) a += w[n] * vbarS[n];
    xS[h] = a;                               // pooled
    xS[HID + h] = ws_q[p * HID + h];         // concat q
  }
  __syncthreads();
#pragma unroll
  for (int cc = 0; cc < 2; ++cc) {
    const int h = t + cc * 256;
    const float* w = W1 + (size_t)h * HID2;
    float a = b1[h];
    for (int j = 0; j < HID2; ++j) a += w[j] * xS[j];
    h1S[h] = fmaxf(a, 0.f);
  }
  __syncthreads();
  if (t < OUT_) {
    const float* w = W2 + (size_t)t * HID;
    float a = b2[t];
    for (int h = 0; h < HID; ++h) a += w[h] * h1S[h];
    logS[t] = a + omask[p * OUT_ + t];
  }
  __syncthreads();
  if (t < OUT_) {
    float mx = logS[0];
    for (int o = 1; o < OUT_; ++o) mx = fmaxf(mx, logS[o]);
    float su = 0.f;
    for (int o = 0; o < OUT_; ++o) su += __expf(logS[o] - mx);
    out[p * OUT_ + t] = __expf(logS[t] - mx) / su;
  }
}

// ---------------------------------------------------------------------------
extern "C" void kernel_launch(void* const* d_in, const int* in_sizes, int n_in,
                              void* d_out, int out_size, void* d_ws, size_t ws_size,
                              hipStream_t stream) {
  (void)in_sizes; (void)n_in; (void)out_size; (void)ws_size;
  const float* video = (const float*)d_in[0];
  const float* ques  = (const float*)d_in[1];
  // d_in[2] = ques_mask: unused by reference
  const float* omask = (const float*)d_in[3];
  const float* Wvp   = (const float*)d_in[4];
  const float* bvp   = (const float*)d_in[5];
  const float* Wqp   = (const float*)d_in[6];
  const float* bqp   = (const float*)d_in[7];
  const float* Wk    = (const float*)d_in[8];
  const float* Wv    = (const float*)d_in[9];
  const float* Wq    = (const float*)d_in[10];
  const float* W1    = (const float*)d_in[11];
  const float* b1    = (const float*)d_in[12];
  const float* W2    = (const float*)d_in[13];
  const float* b2    = (const float*)d_in[14];
  float* out = (float*)d_out;

  char* ws = (char*)d_ws;
  float*    ws_q    = (float*)(ws);                       // 80*512*4   = 163840 B
  float*    ws_ck   = (float*)(ws + 163840);              // 80*512*4   = 163840 B
  float*    ws_part = (float*)(ws + 327680);              // 80*8*520*4 = 1331200 B
  _Float16* w16     = (_Float16*)(ws + 1658880);          // 512*768*2  = 786432 B

  ca_setup<<<NPAIR, 256, 0, stream>>>(ques, Wqp, bqp, Wq, Wk, ws_q, ws_ck);
  ca_cvt<<<(HID * VID_DIM + 255) / 256, 256, 0, stream>>>(Wvp, w16);
  ca_attn<<<NPAIR * NSPLIT, 256, 0, stream>>>(video, w16, bvp, ws_ck, ws_part);
  ca_final<<<NPAIR, 256, 0, stream>>>(ws_part, ws_q, Wv, W1, b1, W2, b2, omask, out);
}
